// WanTransformerBlock_VSA_74277164417728
// MI455X (gfx1250) — matrix operand order repaired and checked by reading, not compiled, not run on hardware
//
// WanTransformerBlock with VSA for MI455X (gfx1250), wave32 + WMMA bf16.
//
// Roofline: ~200 GFLOP, weights read once (~195MB -> ~8us @ 23.3TB/s), so we
// convert f32->bf16 on the fly while staging tiles into LDS and run
// v_wmma_f32_16x16x32_bf16 with f32 accumulation (compute-bound path).

#include <hip/hip_runtime.h>

#define S_LEN   2048
#define L2_LEN  512
#define DIMW    1536
#define NHEADS  12
#define HD      128
#define FFN_D   8960
#define BLKSZ   64
#define NBLK    (S_LEN / BLKSZ)   // 32 coarse blocks
#define TOPK    8
#define EPSF    1e-6f
#define SCALE_F 0.08838834764831845f   // 128^-0.5

typedef __bf16 bf16_t;
typedef bf16_t v16bf __attribute__((ext_vector_type(16)));
typedef float  v8f   __attribute__((ext_vector_type(8)));

static_assert(sizeof(v16bf) == 32, "v16bf must be 32 bytes");

union AFrag { v16bf v; uint4 u[2]; };

__device__ __forceinline__ unsigned short f2bf(float x) {
  unsigned int b = __float_as_uint(x);
  unsigned int r = b + 0x7FFFu + ((b >> 16) & 1u);   // round-to-nearest-even
  return (unsigned short)(r >> 16);
}

__device__ __forceinline__ float gelu_tanh(float x) {
  float x3 = x * x * x;
  return 0.5f * x * (1.0f + tanhf(0.7978845608028654f * (x + 0.044715f * x3)));
}

// ---------------------------------------------------------------------------
// e = scale_shift_table + temb   (6 x 1536)
// ---------------------------------------------------------------------------
__global__ __launch_bounds__(256) void adde_kernel(const float* __restrict__ sst,
                                                   const float* __restrict__ temb,
                                                   float* __restrict__ e) {
  int i = blockIdx.x * 256 + threadIdx.x;     // 0 .. 6*1536-1
  e[i] = sst[i] + temb[i];
}

// ---------------------------------------------------------------------------
// LayerNorm + affine: out = ln(x) * (gplus + gamma[c]) + beta[c]
// One workgroup per row of 1536.
// ---------------------------------------------------------------------------
__global__ __launch_bounds__(256) void ln_kernel(const float* __restrict__ X,
                                                 const float* __restrict__ gamma,
                                                 const float* __restrict__ beta,
                                                 float* __restrict__ Out,
                                                 float gplus) {
  __shared__ float red[256];
  const int row = blockIdx.x;
  const int tid = threadIdx.x;
  const float* x = X + (size_t)row * DIMW;
  float v[6];
  float s = 0.f;
  #pragma unroll
  for (int i = 0; i < 6; ++i) { v[i] = x[tid + i * 256]; s += v[i]; }
  red[tid] = s; __syncthreads();
  for (int off = 128; off > 0; off >>= 1) {
    if (tid < off) red[tid] += red[tid + off];
    __syncthreads();
  }
  const float mean = red[0] * (1.f / DIMW);
  __syncthreads();
  float s2 = 0.f;
  #pragma unroll
  for (int i = 0; i < 6; ++i) { float d = v[i] - mean; s2 += d * d; }
  red[tid] = s2; __syncthreads();
  for (int off = 128; off > 0; off >>= 1) {
    if (tid < off) red[tid] += red[tid + off];
    __syncthreads();
  }
  const float rstd = rsqrtf(red[0] * (1.f / DIMW) + EPSF);
  #pragma unroll
  for (int i = 0; i < 6; ++i) {
    int c = tid + i * 256;
    Out[(size_t)row * DIMW + c] = (v[i] - mean) * rstd * (gplus + gamma[c]) + beta[c];
  }
}

// ---------------------------------------------------------------------------
// RMS norm (over full 1536) + optional RoPE, in place.  One workgroup per row.
// ---------------------------------------------------------------------------
__global__ __launch_bounds__(256) void rmsrope_kernel(float* __restrict__ X,
                                                      const float* __restrict__ wt,
                                                      const float* __restrict__ cosb,
                                                      const float* __restrict__ sinb,
                                                      int do_rope) {
  __shared__ float red[256];
  const int row = blockIdx.x;
  const int tid = threadIdx.x;
  float* x = X + (size_t)row * DIMW;
  float a[3], b[3];
  float s2 = 0.f;
  #pragma unroll
  for (int i = 0; i < 3; ++i) {
    int p = tid + i * 256;                 // pair index 0..767
    a[i] = x[2 * p];
    b[i] = x[2 * p + 1];
    s2 += a[i] * a[i] + b[i] * b[i];
  }
  red[tid] = s2; __syncthreads();
  for (int off = 128; off > 0; off >>= 1) {
    if (tid < off) red[tid] += red[tid + off];
    __syncthreads();
  }
  const float scale = rsqrtf(red[0] * (1.f / DIMW) + EPSF);
  #pragma unroll
  for (int i = 0; i < 3; ++i) {
    int p = tid + i * 256;
    int ii = p & 63;                       // pair within head
    float x1 = a[i] * scale * wt[2 * p];
    float x2 = b[i] * scale * wt[2 * p + 1];
    float o1 = x1, o2 = x2;
    if (do_rope) {
      float c = cosb[row * 64 + ii];
      float sn = sinb[row * 64 + ii];
      o1 = x1 * c - x2 * sn;
      o2 = x2 * c + x1 * sn;
    }
    x[2 * p] = o1;
    x[2 * p + 1] = o2;
  }
}

// ---------------------------------------------------------------------------
// Coarse block means: one workgroup per coarse block (32 blocks of 64 tokens).
// ---------------------------------------------------------------------------
__global__ __launch_bounds__(256) void blockmean_kernel(const float* __restrict__ X,
                                                        float* __restrict__ Outc) {
  const int n = blockIdx.x;
  const int tid = threadIdx.x;
  #pragma unroll
  for (int i = 0; i < 6; ++i) {
    int c = tid + i * 256;
    float s = 0.f;
    for (int r = 0; r < BLKSZ; ++r) s += X[(size_t)(n * BLKSZ + r) * DIMW + c];
    Outc[(size_t)n * DIMW + c] = s * (1.f / BLKSZ);
  }
}

// ---------------------------------------------------------------------------
// Coarse attention per head: pc = softmax(qc kc^T * SCALE), coarse = pc vc,
// plus top-k block indices.  Tiny (32x32x128 per head) -> VALU is fine.
// ---------------------------------------------------------------------------
__global__ __launch_bounds__(256) void coarse_kernel(const float* __restrict__ qc,
                                                     const float* __restrict__ kc,
                                                     const float* __restrict__ vc,
                                                     float* __restrict__ coarse,
                                                     int* __restrict__ blist) {
  __shared__ float qs[NBLK * HD];
  __shared__ float ks[NBLK * HD];
  __shared__ float vs[NBLK * HD];
  __shared__ float sc[NBLK * NBLK];
  __shared__ float scp[NBLK * NBLK];
  const int h = blockIdx.x;
  const int tid = threadIdx.x;
  for (int idx = tid; idx < NBLK * HD; idx += 256) {
    int n = idx / HD, d = idx % HD;
    qs[idx] = qc[(size_t)n * DIMW + h * HD + d];
    ks[idx] = kc[(size_t)n * DIMW + h * HD + d];
    vs[idx] = vc[(size_t)n * DIMW + h * HD + d];
  }
  __syncthreads();
  for (int idx = tid; idx < NBLK * NBLK; idx += 256) {
    int n = idx / NBLK, m = idx % NBLK;
    float s = 0.f;
    for (int d = 0; d < HD; ++d) s += qs[n * HD + d] * ks[m * HD + d];
    sc[idx] = s * SCALE_F;
  }
  __syncthreads();
  if (tid < NBLK) {
    const int n = tid;
    float mx = -1e30f;
    for (int m = 0; m < NBLK; ++m) mx = fmaxf(mx, sc[n * NBLK + m]);
    float sum = 0.f;
    for (int m = 0; m < NBLK; ++m) {
      float p = __expf(sc[n * NBLK + m] - mx);
      sc[n * NBLK + m] = p;
      sum += p;
    }
    float inv = 1.f / sum;
    for (int m = 0; m < NBLK; ++m) {
      float p = sc[n * NBLK + m] * inv;
      sc[n * NBLK + m] = p;
      scp[n * NBLK + m] = p;
    }
    // top-k selection (8 distinct block indices)
    for (int i = 0; i < TOPK; ++i) {
      float best = -1.f; int bj = 0;
      for (int m = 0; m < NBLK; ++m) {
        float v = scp[n * NBLK + m];
        if (v > best) { best = v; bj = m; }
      }
      blist[((size_t)h * NBLK + n) * TOPK + i] = bj;
      scp[n * NBLK + bj] = -1.f;
    }
  }
  __syncthreads();
  for (int idx = tid; idx < NBLK * HD; idx += 256) {
    int n = idx / HD, d = idx % HD;
    float s = 0.f;
    for (int m = 0; m < NBLK; ++m) s += sc[n * NBLK + m] * vs[m * HD + d];
    coarse[(size_t)n * DIMW + h * HD + d] = s;
  }
}

// ---------------------------------------------------------------------------
// Flash attention over 8 key blocks of 64 tokens, WMMA bf16.
// Grid: (head, query_block).  blist==nullptr -> identity blocks (cross attn).
// Q/K/V/O layout: [token][head*128 + d], row stride 1536.
// ---------------------------------------------------------------------------
__global__ __launch_bounds__(256) void flash_kernel(const float* __restrict__ Q,
                                                    const float* __restrict__ Km,
                                                    const float* __restrict__ Vm,
                                                    float* __restrict__ O,
                                                    const int* __restrict__ blist) {
  const int h   = blockIdx.x;
  const int qb  = blockIdx.y;
  const int tid = threadIdx.x;
  const int lane = tid & 31;
  const int w    = tid >> 5;         // wave 0..7
  const int half = (lane >> 4) & 1;
  const int l16  = lane & 15;
  const int mt   = w & 3;            // 16-row tile of the 64 queries
  const int ng   = w >> 2;           // column group

  __shared__ __align__(16) unsigned short stage[128 * 72];  // K:[64][136] or Vt:[128][72]
  __shared__ __align__(16) float          Sbuf[64 * 68];
  __shared__ __align__(16) unsigned short Pbuf[64 * 72];
  __shared__ float m_i[64], l_i[64], al_i[64];

  const int s0 = qb * 64;

  // stage Q (bf16, [row][d], stride 136)
  #pragma unroll
  for (int it = 0; it < 8; ++it) {
    int idx = it * 256 + tid;
    int r = idx >> 5;
    int dq = (idx & 31) << 2;
    const float4 f = *(const float4*)(Q + (size_t)(s0 + r) * DIMW + h * HD + dq);
    unsigned short* p = &stage[r * 136 + dq];
    p[0] = f2bf(f.x); p[1] = f2bf(f.y); p[2] = f2bf(f.z); p[3] = f2bf(f.w);
  }
  __syncthreads();
  // per-wave Q A-fragments for 4 K-steps of 32 over d
  AFrag qf[4];
  {
    const unsigned short* base = &stage[(16 * mt + l16) * 136];
    #pragma unroll
    for (int ks = 0; ks < 4; ++ks) {
      qf[ks].u[0] = *(const uint4*)(base + ks * 32 + 8 * half);
      qf[ks].u[1] = *(const uint4*)(base + ks * 32 + 16 + 8 * half);
    }
  }
  v8f oacc[4];
  #pragma unroll
  for (int j = 0; j < 4; ++j)
    #pragma unroll
    for (int r = 0; r < 8; ++r) oacc[j][r] = 0.f;
  if (tid < 64) { m_i[tid] = -1e30f; l_i[tid] = 0.f; }
  __syncthreads();

  for (int it = 0; it < 8; ++it) {
    const int kb = blist ? blist[((size_t)h * NBLK + qb) * TOPK + it] : it;
    const int t0 = kb * 64;
    // stage K block (bf16, [tok][d], stride 136)
    #pragma unroll
    for (int i2 = 0; i2 < 8; ++i2) {
      int idx = i2 * 256 + tid;
      int r = idx >> 5;
      int dq = (idx & 31) << 2;
      const float4 f = *(const float4*)(Km + (size_t)(t0 + r) * DIMW + h * HD + dq);
      unsigned short* p = &stage[r * 136 + dq];
      p[0] = f2bf(f.x); p[1] = f2bf(f.y); p[2] = f2bf(f.z); p[3] = f2bf(f.w);
    }
    __syncthreads();
    // S = Q K^T  (each wave: 2 n-tiles)
    #pragma unroll
    for (int j = 0; j < 2; ++j) {
      const int nt = ng * 2 + j;
      v8f s;
      #pragma unroll
      for (int r = 0; r < 8; ++r) s[r] = 0.f;
      #pragma unroll
      for (int ks = 0; ks < 4; ++ks) {
        AFrag bf;
        const unsigned short* bb = &stage[(16 * nt + l16) * 136 + ks * 32 + 8 * half];
        bf.u[0] = *(const uint4*)(bb);
        bf.u[1] = *(const uint4*)(bb + 16);
        s = __builtin_amdgcn_wmma_f32_16x16x32_bf16(false, qf[ks].v, false, bf.v,
                                                    (short)0, s, false, false);
      }
      #pragma unroll
      for (int r = 0; r < 8; ++r) {
        int row = 16 * mt + r + 8 * half;
        Sbuf[row * 68 + 16 * nt + l16] = s[r] * SCALE_F;
      }
    }
    __syncthreads();
    // online softmax over this 64-key block
    if (tid < 64) {
      const float* srow = &Sbuf[tid * 68];
      float mold = m_i[tid];
      float mx = mold;
      for (int c = 0; c < 64; ++c) mx = fmaxf(mx, srow[c]);
      float a = __expf(mold - mx);
      float ps = 0.f;
      unsigned short* prow = &Pbuf[tid * 72];
      for (int c = 0; c < 64; ++c) {
        float p = __expf(srow[c] - mx);
        prow[c] = f2bf(p);
        ps += p;
      }
      l_i[tid] = l_i[tid] * a + ps;
      m_i[tid] = mx;
      al_i[tid] = a;
    }
    __syncthreads();
    // stage V transposed (bf16, [d][tok], stride 72)
    #pragma unroll
    for (int i2 = 0; i2 < 8; ++i2) {
      int idx = i2 * 256 + tid;
      int r = idx >> 5;
      int dq = (idx & 31) << 2;
      const float4 f = *(const float4*)(Vm + (size_t)(t0 + r) * DIMW + h * HD + dq);
      stage[(dq + 0) * 72 + r] = f2bf(f.x);
      stage[(dq + 1) * 72 + r] = f2bf(f.y);
      stage[(dq + 2) * 72 + r] = f2bf(f.z);
      stage[(dq + 3) * 72 + r] = f2bf(f.w);
    }
    __syncthreads();
    // rescale O by alpha and accumulate P V
    float alr[8];
    #pragma unroll
    for (int r = 0; r < 8; ++r) alr[r] = al_i[16 * mt + r + 8 * half];
    #pragma unroll
    for (int j = 0; j < 4; ++j)
      #pragma unroll
      for (int r = 0; r < 8; ++r) oacc[j][r] *= alr[r];
    AFrag pf[2];
    {
      const unsigned short* pb = &Pbuf[(16 * mt + l16) * 72];
      #pragma unroll
      for (int ks = 0; ks < 2; ++ks) {
        pf[ks].u[0] = *(const uint4*)(pb + ks * 32 + 8 * half);
        pf[ks].u[1] = *(const uint4*)(pb + ks * 32 + 16 + 8 * half);
      }
    }
    #pragma unroll
    for (int j = 0; j < 4; ++j) {
      const int ntd = ng * 4 + j;
      #pragma unroll
      for (int ks = 0; ks < 2; ++ks) {
        AFrag vf;
        const unsigned short* vb = &stage[(16 * ntd + l16) * 72 + ks * 32 + 8 * half];
        vf.u[0] = *(const uint4*)(vb);
        vf.u[1] = *(const uint4*)(vb + 16);
        oacc[j] = __builtin_amdgcn_wmma_f32_16x16x32_bf16(false, pf[ks].v, false, vf.v,
                                                          (short)0, oacc[j], false, false);
      }
    }
    __syncthreads();
  }
  // epilogue: O / l
  float linv[8];
  #pragma unroll
  for (int r = 0; r < 8; ++r) linv[r] = 1.f / l_i[16 * mt + r + 8 * half];
  #pragma unroll
  for (int j = 0; j < 4; ++j) {
    const int d = 16 * (ng * 4 + j) + l16;
    #pragma unroll
    for (int r = 0; r < 8; ++r) {
      const int row = 16 * mt + r + 8 * half;
      O[(size_t)(s0 + row) * DIMW + h * HD + d] = oacc[j][r] * linv[r];
    }
  }
}

// ---------------------------------------------------------------------------
// Generic bf16 WMMA GEMM: C[M,N] = act(A[M,K] @ W[K,N] + bias[N]).
// BM=128, BN=64, BK=32, 256 threads (8 waves), wave tile 32x32.
// M%128==0, N%64==0, K%32==0 for all call sites.
// ---------------------------------------------------------------------------
__global__ __launch_bounds__(256) void gemm_kernel(const float* __restrict__ A,
                                                   const float* __restrict__ W,
                                                   const float* __restrict__ bias,
                                                   float* __restrict__ C,
                                                   int M, int N, int K, int act) {
  const int tid = threadIdx.x;
  const int lane = tid & 31;
  const int w    = tid >> 5;
  const int half = (lane >> 4) & 1;
  const int l16  = lane & 15;
  const int wm   = w & 3;     // row quarter (32 rows)
  const int wn   = w >> 2;    // col half (32 cols)
  const int n0 = blockIdx.x * 64;
  const int m0 = blockIdx.y * 128;

  __shared__ __align__(16) unsigned short As[128 * 40];
  __shared__ __align__(16) unsigned short Ws[64 * 40];

  v8f acc[2][2];
  #pragma unroll
  for (int i = 0; i < 2; ++i)
    #pragma unroll
    for (int j = 0; j < 2; ++j)
      #pragma unroll
      for (int r = 0; r < 8; ++r) acc[i][j][r] = 0.f;

  const int nk = K >> 5;
  for (int kt = 0; kt < nk; ++kt) {
    const int k0 = kt << 5;
    // A tile 128x32 f32 -> bf16 LDS (stride 40)
    #pragma unroll
    for (int i = 0; i < 4; ++i) {
      int idx = i * 256 + tid;
      int r = idx >> 3;
      int cq = (idx & 7) << 2;
      const float* src = A + (size_t)(m0 + r) * K + k0 + cq;
      if (kt + 1 < nk) __builtin_prefetch(src + 32, 0, 0);
      const float4 f = *(const float4*)src;
      unsigned short* p = &As[r * 40 + cq];
      p[0] = f2bf(f.x); p[1] = f2bf(f.y); p[2] = f2bf(f.z); p[3] = f2bf(f.w);
    }
    // W tile 32x64 f32 -> bf16 LDS transposed [n][k] (stride 40)
    #pragma unroll
    for (int i = 0; i < 2; ++i) {
      int idx = i * 256 + tid;
      int kr = idx >> 4;
      int nq = (idx & 15) << 2;
      const float* src = W + (size_t)(k0 + kr) * N + n0 + nq;
      if (kt + 1 < nk) __builtin_prefetch(src + (size_t)32 * N, 0, 0);
      const float4 f = *(const float4*)src;
      Ws[(nq + 0) * 40 + kr] = f2bf(f.x);
      Ws[(nq + 1) * 40 + kr] = f2bf(f.y);
      Ws[(nq + 2) * 40 + kr] = f2bf(f.z);
      Ws[(nq + 3) * 40 + kr] = f2bf(f.w);
    }
    __syncthreads();
    AFrag af[2], bf[2];
    #pragma unroll
    for (int mtt = 0; mtt < 2; ++mtt) {
      const unsigned short* b = &As[(32 * wm + 16 * mtt + l16) * 40 + 8 * half];
      af[mtt].u[0] = *(const uint4*)(b);
      af[mtt].u[1] = *(const uint4*)(b + 16);
    }
    #pragma unroll
    for (int ntt = 0; ntt < 2; ++ntt) {
      const unsigned short* b = &Ws[(32 * wn + 16 * ntt + l16) * 40 + 8 * half];
      bf[ntt].u[0] = *(const uint4*)(b);
      bf[ntt].u[1] = *(const uint4*)(b + 16);
    }
    #pragma unroll
    for (int mtt = 0; mtt < 2; ++mtt)
      #pragma unroll
      for (int ntt = 0; ntt < 2; ++ntt)
        acc[mtt][ntt] = __builtin_amdgcn_wmma_f32_16x16x32_bf16(
            false, af[mtt].v, false, bf[ntt].v, (short)0, acc[mtt][ntt], false, false);
    __syncthreads();
  }
  // epilogue
  #pragma unroll
  for (int mtt = 0; mtt < 2; ++mtt)
    #pragma unroll
    for (int ntt = 0; ntt < 2; ++ntt) {
      const int col = n0 + 32 * wn + 16 * ntt + l16;
      const float bv = bias[col];
      #pragma unroll
      for (int r = 0; r < 8; ++r) {
        const int row = m0 + 32 * wm + 16 * mtt + r + 8 * half;
        float v = acc[mtt][ntt][r] + bv;
        if (act == 1) v = gelu_tanh(v);
        C[(size_t)row * N + col] = v;
      }
    }
}

// ---------------------------------------------------------------------------
// fine += g * coarse[blk(s)]
// ---------------------------------------------------------------------------
__global__ __launch_bounds__(256) void combine_kernel(float* __restrict__ fine,
                                                      const float* __restrict__ g,
                                                      const float* __restrict__ coarse) {
  size_t i = (size_t)blockIdx.x * 256 + threadIdx.x;  // over S_LEN*DIMW
  int s = (int)(i / DIMW);
  int c = (int)(i % DIMW);
  fine[i] += g[i] * coarse[(size_t)(s >> 6) * DIMW + c];
}

// ---------------------------------------------------------------------------
// out = a + gate[c] * b   (gate==nullptr -> plain add)
// ---------------------------------------------------------------------------
__global__ __launch_bounds__(256) void resid_kernel(const float* __restrict__ a,
                                                    const float* __restrict__ b,
                                                    const float* __restrict__ gate,
                                                    float* __restrict__ out) {
  size_t i = (size_t)blockIdx.x * 256 + threadIdx.x;
  int c = (int)(i % DIMW);
  float gv = gate ? gate[c] : 1.f;
  out[i] = a[i] + gv * b[i];
}

// ---------------------------------------------------------------------------
// Host orchestration
// ---------------------------------------------------------------------------
extern "C" void kernel_launch(void* const* d_in, const int* in_sizes, int n_in,
                              void* d_out, int out_size, void* d_ws, size_t ws_size,
                              hipStream_t stream) {
  (void)in_sizes; (void)n_in; (void)out_size; (void)ws_size;
  const float* hs    = (const float*)d_in[0];
  const float* enc   = (const float*)d_in[1];
  const float* temb  = (const float*)d_in[2];
  const float* cosb  = (const float*)d_in[3];
  const float* sinb  = (const float*)d_in[4];
  const float* sst   = (const float*)d_in[5];
  const float* Wq  = (const float*)d_in[6];  const float* bq  = (const float*)d_in[7];
  const float* Wk  = (const float*)d_in[8];  const float* bk  = (const float*)d_in[9];
  const float* Wv  = (const float*)d_in[10]; const float* bv  = (const float*)d_in[11];
  const float* Wg  = (const float*)d_in[12]; const float* bg  = (const float*)d_in[13];
  const float* Wo  = (const float*)d_in[14]; const float* bo  = (const float*)d_in[15];
  const float* qnw = (const float*)d_in[16]; const float* knw = (const float*)d_in[17];
  const float* salnw = (const float*)d_in[18]; const float* salnb = (const float*)d_in[19];
  const float* cWq = (const float*)d_in[20]; const float* cbq = (const float*)d_in[21];
  const float* cWk = (const float*)d_in[22]; const float* cbk = (const float*)d_in[23];
  const float* cWv = (const float*)d_in[24]; const float* cbv = (const float*)d_in[25];
  const float* cWo = (const float*)d_in[26]; const float* cbo = (const float*)d_in[27];
  const float* cqnw = (const float*)d_in[28]; const float* cknw = (const float*)d_in[29];
  const float* W1 = (const float*)d_in[30]; const float* b1 = (const float*)d_in[31];
  const float* W2 = (const float*)d_in[32]; const float* b2 = (const float*)d_in[33];

  // workspace partition (deterministic static layout)
  char* p = (char*)d_ws;
  auto alloc = [&](size_t nbytes) -> void* {
    void* r = (void*)p;
    p += (nbytes + 255) & ~(size_t)255;
    return r;
  };
  const size_t NS = (size_t)S_LEN * DIMW * sizeof(float);       // 12.6 MB
  float* e      = (float*)alloc(6 * DIMW * sizeof(float));
  float* normb  = (float*)alloc(NS);
  float* qb     = (float*)alloc(NS);
  float* kb_    = (float*)alloc(NS);
  float* vb_    = (float*)alloc(NS);
  float* gb     = (float*)alloc(NS);
  float* fineb  = (float*)alloc(NS);
  float* tmpb   = (float*)alloc(NS);
  float* resb   = (float*)alloc(NS);
  float* res2b  = (float*)alloc(NS);
  float* ckb    = (float*)alloc((size_t)L2_LEN * DIMW * sizeof(float));
  float* cvb    = (float*)alloc((size_t)L2_LEN * DIMW * sizeof(float));
  float* h1b    = (float*)alloc((size_t)S_LEN * FFN_D * sizeof(float));  // 73 MB
  float* qcb    = (float*)alloc((size_t)NBLK * DIMW * sizeof(float));
  float* kcb    = (float*)alloc((size_t)NBLK * DIMW * sizeof(float));
  float* vcb    = (float*)alloc((size_t)NBLK * DIMW * sizeof(float));
  float* coarseb= (float*)alloc((size_t)NBLK * DIMW * sizeof(float));
  int*   blist  = (int*)alloc((size_t)NHEADS * NBLK * TOPK * sizeof(int));

  const int EW = (S_LEN * DIMW) / 256;   // elementwise grid over 2048x1536
  const dim3 gDD(DIMW / 64, S_LEN / 128);      // (24,16) 2048x1536 GEMM
  const dim3 gED(DIMW / 64, L2_LEN / 128);     // (24,4)  512x1536 GEMM
  const dim3 gFF(FFN_D / 64, S_LEN / 128);     // (140,16) FFN up

  // modulation
  adde_kernel<<<(6 * DIMW) / 256, 256, 0, stream>>>(sst, temb, e);
  // norm_hs = ln(hs)*(1+scale_msa)+shift_msa
  ln_kernel<<<S_LEN, 256, 0, stream>>>(hs, e + DIMW, e, normb, 1.0f);
  // q/k/v/g projections
  gemm_kernel<<<gDD, 256, 0, stream>>>(normb, Wq, bq, qb,  S_LEN, DIMW, DIMW, 0);
  gemm_kernel<<<gDD, 256, 0, stream>>>(normb, Wk, bk, kb_, S_LEN, DIMW, DIMW, 0);
  gemm_kernel<<<gDD, 256, 0, stream>>>(normb, Wv, bv, vb_, S_LEN, DIMW, DIMW, 0);
  gemm_kernel<<<gDD, 256, 0, stream>>>(normb, Wg, bg, gb,  S_LEN, DIMW, DIMW, 0);
  // rms + rope on q,k (in place)
  rmsrope_kernel<<<S_LEN, 256, 0, stream>>>(qb,  qnw, cosb, sinb, 1);
  rmsrope_kernel<<<S_LEN, 256, 0, stream>>>(kb_, knw, cosb, sinb, 1);
  // coarse path
  blockmean_kernel<<<NBLK, 256, 0, stream>>>(qb,  qcb);
  blockmean_kernel<<<NBLK, 256, 0, stream>>>(kb_, kcb);
  blockmean_kernel<<<NBLK, 256, 0, stream>>>(vb_, vcb);
  coarse_kernel<<<NHEADS, 256, 0, stream>>>(qcb, kcb, vcb, coarseb, blist);
  // sparse fine attention (8 top-k key blocks per query block)
  flash_kernel<<<dim3(NHEADS, NBLK), 256, 0, stream>>>(qb, kb_, vb_, fineb, blist);
  // attn = fine + g*coarse ; project
  combine_kernel<<<EW, 256, 0, stream>>>(fineb, gb, coarseb);
  gemm_kernel<<<gDD, 256, 0, stream>>>(fineb, Wo, bo, tmpb, S_LEN, DIMW, DIMW, 0);
  resid_kernel<<<EW, 256, 0, stream>>>(hs, tmpb, e + 2 * DIMW, resb);
  // cross attention
  ln_kernel<<<S_LEN, 256, 0, stream>>>(resb, salnw, salnb, normb, 0.0f);
  gemm_kernel<<<gDD, 256, 0, stream>>>(normb, cWq, cbq, qb, S_LEN, DIMW, DIMW, 0);
  rmsrope_kernel<<<S_LEN, 256, 0, stream>>>(qb, cqnw, cosb, sinb, 0);
  gemm_kernel<<<gED, 256, 0, stream>>>(enc, cWk, cbk, ckb, L2_LEN, DIMW, DIMW, 0);
  rmsrope_kernel<<<L2_LEN, 256, 0, stream>>>(ckb, cknw, cosb, sinb, 0);
  gemm_kernel<<<gED, 256, 0, stream>>>(enc, cWv, cbv, cvb, L2_LEN, DIMW, DIMW, 0);
  flash_kernel<<<dim3(NHEADS, NBLK), 256, 0, stream>>>(qb, ckb, cvb, fineb, (const int*)nullptr);
  gemm_kernel<<<gDD, 256, 0, stream>>>(fineb, cWo, cbo, tmpb, S_LEN, DIMW, DIMW, 0);
  resid_kernel<<<EW, 256, 0, stream>>>(resb, tmpb, (const float*)nullptr, res2b);
  // FFN
  ln_kernel<<<S_LEN, 256, 0, stream>>>(res2b, e + 4 * DIMW, e + 3 * DIMW, normb, 1.0f);
  gemm_kernel<<<gFF, 256, 0, stream>>>(normb, W1, b1, h1b, S_LEN, FFN_D, DIMW, 1);
  gemm_kernel<<<gDD, 256, 0, stream>>>(h1b, W2, b2, tmpb, S_LEN, DIMW, FFN_D, 0);
  resid_kernel<<<EW, 256, 0, stream>>>(res2b, tmpb, e + 5 * DIMW, (float*)d_out);
}
